// LSTM_VAE_71090298683871
// MI455X (gfx1250) — matrix operand order repaired and checked by reading, not compiled, not run on hardware
//
#include <hip/hip_runtime.h>

typedef _Float16 v8h  __attribute__((ext_vector_type(8)));
typedef _Float16 v16h __attribute__((ext_vector_type(16)));
typedef float    v8f  __attribute__((ext_vector_type(8)));

#define B_   4096
#define T_   512
#define I_   25
#define H_   20
#define L_   10
#define TI   (T_ * I_)
#define RSTR 72           // halves per row in LDS tiles (16B-aligned rows, conflict-free)
#define RPB  16           // batch rows per block (1 wave x 16)

// d_out flat offsets (floats), reference return order
#define MEAN_OFF 52428800L
#define LV_OFF   52469760L
#define Z_OFF    52510720L
#define HN_OFF   52551680L
#define CN_OFF   52633600L

#if __has_builtin(__builtin_amdgcn_tanhf)
__device__ __forceinline__ float tanh_fast(float x) { return __builtin_amdgcn_tanhf(x); }
#elif __has_builtin(__builtin_amdgcn_tanh_f32)
__device__ __forceinline__ float tanh_fast(float x) { return __builtin_amdgcn_tanh_f32(x); }
#else
__device__ __forceinline__ float tanh_fast(float x) { return 2.0f / (1.0f + __expf(-2.0f * x)) - 1.0f; }
#endif
__device__ __forceinline__ float sigm(float x) { return 0.5f * tanh_fast(0.5f * x) + 0.5f; }

__device__ __forceinline__ v8f splat8(float v) {
  v8f r;
#pragma unroll
  for (int i = 0; i < 8; ++i) r[i] = v;
  return r;
}

__device__ __forceinline__ v16h cat16(v8h lo, v8h hi) {
  return __builtin_shufflevector(lo, hi, 0, 1, 2, 3, 4, 5, 6, 7, 8, 9, 10, 11, 12, 13, 14, 15);
}

// A fragment (16x32 f16): lane holds K = ks*32 + {8*half..+7} and {16+8*half..+7}
__device__ __forceinline__ v16h load_a(const _Float16* row, int ks, int half) {
  const _Float16* p = row + ks * 32 + 8 * half;
  return cat16(*(const v8h*)p, *(const v8h*)(p + 16));
}
// B fragment (32x16 f16): lane holds K = ks*32 + {8*half..+7} and {16+8*half..+7} for column n
__device__ __forceinline__ v16h load_b(const _Float16* W, int nrow, int ks, int half) {
  const _Float16* p = W + nrow * RSTR + ks * 32 + 8 * half;
  return cat16(*(const v8h*)p, *(const v8h*)(p + 16));
}

// One LSTM over T steps for the 16-row batch tile owned by this (single) wave.
// Wg: LDS f16 [128 padded-gate-cols][RSTR] (K: 0..24 x | 32..51 h, rest 0)
// Wo: LDS f16 [32][RSTR] (proj weights, K=j 0..19, rest 0)
// xh: LDS f16 [2 buffers][16 rows][RSTR]: cols 0..31 = x(t), 32..63 = h(t-1)
// hs32/cs32: LDS f32 [16][20] phase-boundary h/c
template <bool PROJ>
__device__ void run_lstm(const float* __restrict__ x,
                         const _Float16* __restrict__ Wg,
                         const _Float16* __restrict__ Wo,
                         _Float16* __restrict__ xh,
                         float* __restrict__ hs32, float* __restrict__ cs32,
                         const float* __restrict__ bih, const float* __restrict__ bhh,
                         const float* __restrict__ outb, float* __restrict__ xhat,
                         long gRow0, int lane)
{
  const int half = lane >> 4;
  const int r    = lane & 15;
  const float* xbase = x + gRow0 * (long)TI;

  // per-lane bias fragments for the 8 padded N-tiles (tile 2g -> j=r, 2g+1 -> j=16+r)
  float bb[8];
#pragma unroll
  for (int g4 = 0; g4 < 4; ++g4) {
    int c0 = g4 * H_ + r;
    bb[2 * g4]     = bih[c0] + bhh[c0];
    bb[2 * g4 + 1] = (r < 4) ? (bih[g4 * H_ + 16 + r] + bhh[g4 * H_ + 16 + r]) : 0.f;
  }
  float pb[2] = {0.f, 0.f};
  if (PROJ) {
    pb[0] = outb[r];
    pb[1] = (16 + r < I_) ? outb[16 + r] : 0.f;
  }

  // cache all B fragments in registers for the whole phase
  v16h Bg[2][8];
#pragma unroll
  for (int ks = 0; ks < 2; ++ks)
#pragma unroll
    for (int n = 0; n < 8; ++n)
      Bg[ks][n] = load_b(Wg, n * 16 + r, ks, half);
  v16h Bo[2];
  if (PROJ) {
#pragma unroll
    for (int n = 0; n < 2; ++n)
      Bo[n] = load_b(Wo, n * 16 + r, 0, half);
  }

  // prologue: stage x(0) and h(init) (zero-padded) into buffer 0; init c fragments
#pragma unroll
  for (int rr = 0; rr < 16; ++rr) {
    float v  = (lane < I_) ? __builtin_nontemporal_load(&xbase[(long)rr * TI + lane]) : 0.f;
    xh[rr * RSTR + lane] = (_Float16)v;
    float hv = (lane < H_) ? hs32[rr * H_ + lane] : 0.f;
    xh[rr * RSTR + 32 + lane] = (_Float16)hv;
  }
  float cA[8], cB[8], hA[8], hB[8];
#pragma unroll
  for (int d = 0; d < 8; ++d) {
    int row = d + 8 * half;
    cA[d] = cs32[row * H_ + r];
    cB[d] = (r < 4) ? cs32[row * H_ + 16 + r] : 0.f;
    hA[d] = 0.f; hB[d] = 0.f;
  }
  v16h a1 = load_a(xh + r * RSTR, 1, half);

  for (int t = 0; t < T_; ++t) {
    _Float16* xb  = xh + (t & 1) * (16 * RSTR);
    _Float16* xbn = xh + ((t + 1) & 1) * (16 * RSTR);

    // prefetch x(t+1) into registers (off the critical path); NT: pure stream
    int tn = (t + 1 < T_) ? t + 1 : t;
    const float* xtn = xbase + (long)tn * I_;
    float xp[16];
#pragma unroll
    for (int rr = 0; rr < 16; ++rr)
      xp[rr] = (lane < I_) ? __builtin_nontemporal_load(&xtn[(long)rr * TI + lane]) : 0.f;
    // warm the cache two steps ahead (global_prefetch_b8)
    {
      int t2 = (t + 2 < T_) ? t + 2 : t;
      __builtin_prefetch(xbase + (long)r * TI + (long)t2 * I_ + half * 16, 0, 1);
    }

    // gates = bias + x@Wx + h@Wh : 16 x v_wmma_f32_16x16x32_f16, h-chain depth 1
    v8f acc[8];
#pragma unroll
    for (int n = 0; n < 8; ++n) acc[n] = splat8(bb[n]);
    v16h a0 = load_a(xb + r * RSTR, 0, half);
#pragma unroll
    for (int n = 0; n < 8; ++n)
      acc[n] = __builtin_amdgcn_wmma_f32_16x16x32_f16(false, a0, false, Bg[0][n],
                                                      (short)0, acc[n], false, false);
#pragma unroll
    for (int n = 0; n < 8; ++n)
      acc[n] = __builtin_amdgcn_wmma_f32_16x16x32_f16(false, a1, false, Bg[1][n],
                                                      (short)0, acc[n], false, false);

    // in-register LSTM cell (gate g for (row,j) sits in acc[2g+hi][d] of THIS lane)
#pragma unroll
    for (int d = 0; d < 8; ++d) {
      int row = d + 8 * half;
      float iv = sigm(acc[0][d]);
      float fv = sigm(acc[2][d]);
      float gv = tanh_fast(acc[4][d]);
      float ov = sigm(acc[6][d]);
      cA[d] = fv * cA[d] + iv * gv;
      hA[d] = ov * tanh_fast(cA[d]);
      xbn[row * RSTR + 32 + r] = (_Float16)hA[d];

      float iv2 = sigm(acc[1][d]);
      float fv2 = sigm(acc[3][d]);
      float gv2 = tanh_fast(acc[5][d]);
      float ov2 = sigm(acc[7][d]);
      cB[d] = fv2 * cB[d] + iv2 * gv2;
      hB[d] = ov2 * tanh_fast(cB[d]);
      xbn[row * RSTR + 48 + r] = (_Float16)hB[d];   // pads killed by zero weight cols
    }
    // commit the prefetched x tile to the next buffer
#pragma unroll
    for (int rr = 0; rr < 16; ++rr)
      xbn[rr * RSTR + lane] = (_Float16)xp[rr];

    // reload recurrent A fragment = h(t); also feeds the decoder projection
    a1 = load_a(xbn + r * RSTR, 1, half);

    if (PROJ) {
      v8f po[2] = { splat8(pb[0]), splat8(pb[1]) };
#pragma unroll
      for (int n = 0; n < 2; ++n)
        po[n] = __builtin_amdgcn_wmma_f32_16x16x32_f16(false, a1, false, Bo[n],
                                                       (short)0, po[n], false, false);
      float* xo = xhat + gRow0 * (long)TI + (long)t * I_;
#pragma unroll
      for (int n = 0; n < 2; ++n) {
        int col = n * 16 + r;
        if (col < I_) {
#pragma unroll
          for (int d = 0; d < 8; ++d)
            __builtin_nontemporal_store(po[n][d], &xo[(long)(d + 8 * half) * TI + col]);
        }
      }
    }
  }

  if (!PROJ) {  // encoder: spill f32 h_n/c_n for the VAE head
#pragma unroll
    for (int d = 0; d < 8; ++d) {
      int row = d + 8 * half;
      hs32[row * H_ + r] = hA[d];
      cs32[row * H_ + r] = cA[d];
      if (r < 4) {
        hs32[row * H_ + 16 + r] = hB[d];
        cs32[row * H_ + 16 + r] = cB[d];
      }
    }
  }
}

__global__ __launch_bounds__(32)
void lstm_vae_kernel(const float* __restrict__ x,
                     const float* __restrict__ h0, const float* __restrict__ c0,
                     const float* __restrict__ noise,
                     const float* __restrict__ eWih, const float* __restrict__ eWhh,
                     const float* __restrict__ ebih, const float* __restrict__ ebhh,
                     const float* __restrict__ dWih, const float* __restrict__ dWhh,
                     const float* __restrict__ dbih, const float* __restrict__ dbhh,
                     const float* __restrict__ meanW, const float* __restrict__ meanb,
                     const float* __restrict__ lvW, const float* __restrict__ lvb,
                     const float* __restrict__ initW, const float* __restrict__ initb,
                     const float* __restrict__ outW, const float* __restrict__ outb,
                     float* __restrict__ out)
{
  __shared__ __align__(16) _Float16 sWe[128 * RSTR];
  __shared__ __align__(16) _Float16 sWd[128 * RSTR];
  __shared__ __align__(16) _Float16 sWo[32 * RSTR];
  __shared__ __align__(16) _Float16 sXH[2 * 16 * RSTR];
  __shared__ float sH32[16 * H_];
  __shared__ float sC32[16 * H_];

  const int lane = threadIdx.x;
  const long gRow0 = (long)blockIdx.x * RPB;

  // ---- stage f16 weights: padded-gate layout [n^ = g*32+jj][K 0..63] ----
  for (int idx = lane; idx < 128 * 64; idx += 32) {
    int nn = idx >> 6, k = idx & 63;
    int g4 = nn >> 5, jj = nn & 31;
    int c = g4 * H_ + jj;
    float ve = 0.f, vd = 0.f;
    if (jj < H_) {
      if (k < I_)                    { ve = eWih[c * I_ + k];        vd = dWih[c * I_ + k]; }
      else if (k >= 32 && k < 32+H_) { ve = eWhh[c * H_ + (k - 32)]; vd = dWhh[c * H_ + (k - 32)]; }
    }
    sWe[nn * RSTR + k] = (_Float16)ve;
    sWd[nn * RSTR + k] = (_Float16)vd;
  }
  for (int idx = lane; idx < 32 * 32; idx += 32) {
    int nn = idx >> 5, k = idx & 31;
    float v = (nn < I_ && k < H_) ? outW[nn * H_ + k] : 0.f;
    sWo[nn * RSTR + k] = (_Float16)v;
  }
  // ---- init h, c from h0/c0 ----
  for (int p = lane; p < 16 * H_; p += 32) {
    sH32[p] = h0[gRow0 * H_ + p];
    sC32[p] = c0[gRow0 * H_ + p];
  }
  __syncthreads();   // single-wave workgroup: lowers to S_NOP, kept for safety

  // ================= encoder =================
  run_lstm<false>(x, sWe, nullptr, sXH, sH32, sC32,
                  ebih, ebhh, nullptr, nullptr, gRow0, lane);

  for (int p = lane; p < 16 * H_; p += 32) {
    out[HN_OFF + gRow0 * H_ + p] = sH32[p];
    out[CN_OFF + gRow0 * H_ + p] = sC32[p];
  }
  __syncthreads();

  // ============ VAE head: mean / log_var / z / decoder init ============
  if (lane < RPB) {
    long row = gRow0 + lane;
    float hv[H_];
#pragma unroll
    for (int j = 0; j < H_; ++j) hv[j] = sH32[lane * H_ + j];
    float zv[L_];
#pragma unroll
    for (int l = 0; l < L_; ++l) {
      float m = meanb[l], lv = lvb[l];
#pragma unroll
      for (int j = 0; j < H_; ++j) {
        m  += hv[j] * meanW[l * H_ + j];
        lv += hv[j] * lvW[l * H_ + j];
      }
      float zz = noise[row * L_ + l] * __expf(0.5f * lv) + m;
      out[MEAN_OFF + row * L_ + l] = m;
      out[LV_OFF   + row * L_ + l] = lv;
      out[Z_OFF    + row * L_ + l] = zz;
      zv[l] = zz;
    }
#pragma unroll
    for (int j = 0; j < H_; ++j) {
      float hd = initb[j];
#pragma unroll
      for (int l = 0; l < L_; ++l) hd += zv[l] * initW[j * L_ + l];
      sH32[lane * H_ + j] = hd;
      sC32[lane * H_ + j] = hd;
    }
  }
  __syncthreads();

  // ================= decoder (+ fused output projection) =================
  run_lstm<true>(x, sWd, sWo, sXH, sH32, sC32,
                 dbih, dbhh, outb, out, gRow0, lane);
}

extern "C" void kernel_launch(void* const* d_in, const int* in_sizes, int n_in,
                              void* d_out, int out_size, void* d_ws, size_t ws_size,
                              hipStream_t stream) {
  const float* x     = (const float*)d_in[0];
  const float* h0    = (const float*)d_in[1];
  const float* c0    = (const float*)d_in[2];
  const float* noise = (const float*)d_in[3];
  const float* eWih  = (const float*)d_in[4];
  const float* eWhh  = (const float*)d_in[5];
  const float* ebih  = (const float*)d_in[6];
  const float* ebhh  = (const float*)d_in[7];
  const float* dWih  = (const float*)d_in[8];
  const float* dWhh  = (const float*)d_in[9];
  const float* dbih  = (const float*)d_in[10];
  const float* dbhh  = (const float*)d_in[11];
  const float* meanW = (const float*)d_in[12];
  const float* meanb = (const float*)d_in[13];
  const float* lvW   = (const float*)d_in[14];
  const float* lvb   = (const float*)d_in[15];
  const float* initW = (const float*)d_in[16];
  const float* initb = (const float*)d_in[17];
  const float* outW  = (const float*)d_in[18];
  const float* outb  = (const float*)d_in[19];
  float* out = (float*)d_out;

  dim3 grid(B_ / RPB);   // 256 blocks: one recurrent chain per wave, one wave per block
  dim3 block(32);
  lstm_vae_kernel<<<grid, block, 0, stream>>>(
      x, h0, c0, noise, eWih, eWhh, ebih, ebhh,
      dWih, dWhh, dbih, dbhh, meanW, meanb, lvW, lvb,
      initW, initb, outW, outb, out);
}
